// AttentionMapGenerator_67843303407689
// MI455X (gfx1250) — matrix operand order repaired and checked by reading, not compiled, not run on hardware
//
#include <hip/hip_runtime.h>
#include <hip/hip_bf16.h>

typedef __attribute__((ext_vector_type(16))) _Float16 v16h;
typedef __attribute__((ext_vector_type(8)))  _Float16 v8h;
typedef __attribute__((ext_vector_type(8)))  float    v8f;

#define LRELU_SLOPE 0.2f
#define BN_EPS 1e-5f

// ---------------------------------------------------------------------------
// Weight transpose + f32->f16 convert:  Wt[n][k] = (f16) W[k][n]
// ---------------------------------------------------------------------------
__global__ __launch_bounds__(256) void wtrans_kernel(
    const float* __restrict__ W, _Float16* __restrict__ Wt, int K, int N) {
  long i = (long)blockIdx.x * blockDim.x + threadIdx.x;
  if (i >= (long)K * N) return;
  int k = (int)(i % K);
  long n = i / K;
  Wt[i] = (_Float16)W[(long)k * N + n];
}

// Plain f32 -> f16 convert (for the raw network input)
__global__ __launch_bounds__(256) void cvt16_kernel(
    const float* __restrict__ in, _Float16* __restrict__ out, long n) {
  long i = (long)blockIdx.x * blockDim.x + threadIdx.x;
  if (i < n) out[i] = (_Float16)in[i];
}

// ---------------------------------------------------------------------------
// Fused gather-GEMM via WMMA, f16 inputs, f32 accumulate, N-blocked by NT.
//   If neigh != null: A[m][j*C+c] = X[b, neigh[n*7+j], c]  (m = b*N + n)
//   else:             A[m][c]     = X[m*C + c]             (plain GEMM, K = C)
//   Wt is Cout x K row-major f16 (pre-transposed weights), K = nj*C.
//   One wave computes a 16 x (16*NT) output tile.
//   Requires: M%16==0, Cout%(16*NT)==0, C%32==0.
// ---------------------------------------------------------------------------
template <int NT>
__global__ __launch_bounds__(32) void wmma_gconv_kernel(
    const _Float16* __restrict__ X, const int* __restrict__ neigh,
    const _Float16* __restrict__ Wt, const float* __restrict__ bias,
    float* __restrict__ Out, int N, int C, int Cout) {
  const int lane = threadIdx.x;
  const int r = lane & 15;
  const int hi = lane >> 4;
  const int m = blockIdx.x * 16 + r;            // A-frag row / D rows
  const int col0 = blockIdx.y * (16 * NT) + r;  // first B-frag column

  const int nj = neigh ? 7 : 1;
  const long K = (long)nj * C;

  int b = 0, n = 0;
  if (neigh) { b = m / N; n = m - b * N; }

  v8f acc[NT];
#pragma unroll
  for (int t = 0; t < NT; ++t) acc[t] = (v8f){};

  for (int j = 0; j < nj; ++j) {
    long rb;
    if (neigh) rb = ((long)b * N + neigh[n * 7 + j]) * (long)C;
    else       rb = (long)m * C;
    const _Float16* xr = X + rb;

    for (int c0 = 0; c0 < C; c0 += 32) {
      // A fragment (ISA 16-bit A 16x32 layout):
      //   lane half hi=0: K {c0+0..7, c0+16..23}; hi=1: K {c0+8..15, c0+24..31}
      v16h a;
      {
        v8h a0 = *(const v8h*)(xr + c0 + hi * 8);
        v8h a1 = *(const v8h*)(xr + c0 + 16 + hi * 8);
#pragma unroll
        for (int q = 0; q < 8; ++q) { a[q] = a0[q]; a[8 + q] = a1[q]; }
      }
      // B fragments (ISA 16-bit B 32x16 layout):
      //   lanes 0-15 hold K {c0+0..7, c0+16..23} of their column; lanes 16-31 hold {c0+8..15, c0+24..31}
#pragma unroll
      for (int t = 0; t < NT; ++t) {
        const _Float16* wr =
            Wt + (long)(col0 + 16 * t) * K + (long)j * C + c0 + hi * 8;
        v8h w0 = *(const v8h*)(wr);
        v8h w1 = *(const v8h*)(wr + 16);
        v16h bf;
#pragma unroll
        for (int q = 0; q < 8; ++q) { bf[q] = w0[q]; bf[8 + q] = w1[q]; }
        acc[t] = __builtin_amdgcn_wmma_f32_16x16x32_f16(
            false, a, false, bf, (short)0, acc[t], false, false);
      }
    }
  }

  // D layout: lane holds column (lane&15); rows (hi*8 + i) of the tile.
  const long row0 = (long)blockIdx.x * 16 + hi * 8;
#pragma unroll
  for (int t = 0; t < NT; ++t) {
    int col = col0 + 16 * t;
    float bv = bias ? bias[col] : 0.0f;
#pragma unroll
    for (int i = 0; i < 8; ++i)
      Out[(row0 + i) * (long)Cout + col] = acc[t][i] + bv;
  }
}

// ---------------------------------------------------------------------------
// Deterministic per-channel sum / sum-of-squares partials:  P[2][S][C]
// ---------------------------------------------------------------------------
__global__ __launch_bounds__(256) void col_partial_kernel(
    const float* __restrict__ X, long M, int C, float* __restrict__ P, int S) {
  int c = blockIdx.x;
  int s = blockIdx.y;
  long per = (M + S - 1) / S;
  long r0 = (long)s * per;
  long r1 = r0 + per; if (r1 > M) r1 = M;
  float sum = 0.f, sq = 0.f;
  for (long r = r0 + threadIdx.x; r < r1; r += 256) {
    float v = X[r * (long)C + c];
    sum += v; sq += v * v;
  }
  __shared__ float sh0[256], sh1[256];
  sh0[threadIdx.x] = sum; sh1[threadIdx.x] = sq;
  __syncthreads();
  for (int o = 128; o > 0; o >>= 1) {
    if ((int)threadIdx.x < o) {
      sh0[threadIdx.x] += sh0[threadIdx.x + o];
      sh1[threadIdx.x] += sh1[threadIdx.x + o];
    }
    __syncthreads();
  }
  if (threadIdx.x == 0) {
    P[(long)s * C + c] = sh0[0];
    P[(long)S * C + (long)s * C + c] = sh1[0];
  }
}

// Finalize: ss[c] = g*rsqrt(var+eps), ss[C+c] = be - mean*scale
__global__ void col_finalize_kernel(
    const float* __restrict__ P, int S, int C, long M,
    const float* __restrict__ g, const float* __restrict__ be,
    float* __restrict__ ss) {
  int c = blockIdx.x * blockDim.x + threadIdx.x;
  if (c >= C) return;
  float sum = 0.f, sq = 0.f;
  for (int s = 0; s < S; ++s) {
    sum += P[(long)s * C + c];
    sq  += P[(long)S * C + (long)s * C + c];
  }
  float mean = sum / (float)M;
  float var = sq / (float)M - mean * mean;
  float sc = g[c] * rsqrtf(var + BN_EPS);
  ss[c] = sc;
  ss[C + c] = be[c] - mean * sc;
}

// BN + leaky-relu (+ optional sigmoid); writes f32 (outf) or f16 (outh).
__global__ __launch_bounds__(256) void bn_act_kernel(
    const float* __restrict__ in, float* __restrict__ outf,
    _Float16* __restrict__ outh, long total, int C,
    const float* __restrict__ ss, int sigm) {
  long i = (long)blockIdx.x * blockDim.x + threadIdx.x;
  if (i >= total) return;
  int c = (int)(i % C);
  float v = in[i] * ss[c] + ss[C + c];
  v = (v >= 0.f) ? v : LRELU_SLOPE * v;
  if (sigm) v = 1.0f / (1.0f + __expf(-v));
  if (outh) outh[i] = (_Float16)v;
  else      outf[i] = v;
}

// ---------------------------------------------------------------------------
// Gathered GEMV (Cout == 1): one wave per output row, wave shuffle reduce.
//   out[m] = sum_{j,c} H[b, neigh[n,j], c] * w[j*ch + c] + bias[0]
// ---------------------------------------------------------------------------
__global__ __launch_bounds__(256) void dot1_kernel(
    const float* __restrict__ H, const int* __restrict__ neigh,
    const float* __restrict__ w, const float* __restrict__ bias,
    float* __restrict__ out, int N, int ch, int lch) {
  int wid = threadIdx.x >> 5;
  int lane = threadIdx.x & 31;
  long m = (long)blockIdx.x * 8 + wid;
  int b = (int)(m / N);
  int n = (int)(m - (long)b * N);
  float acc = 0.f;
  int K = 7 * ch;
  for (int k = lane; k < K; k += 32) {
    int j = k >> lch;
    int c = k & (ch - 1);
    acc += H[((long)b * N + neigh[n * 7 + j]) * (long)ch + c] * w[k];
  }
#pragma unroll
  for (int o = 16; o > 0; o >>= 1) acc += __shfl_xor(acc, o, 32);
  if (lane == 0) out[m] = acc + bias[0];
}

// ---------------------------------------------------------------------------
// Upconv select: x1 = y[:, top]; x2 = mean of down-pairs; concat. f16 output.
// ---------------------------------------------------------------------------
__global__ __launch_bounds__(256) void upsel_kernel(
    const float* __restrict__ Y, const int* __restrict__ top,
    const int* __restrict__ down, _Float16* __restrict__ Xo,
    int B, int Nraw, int Nnew, int cout) {
  long i = (long)blockIdx.x * blockDim.x + threadIdx.x;
  long tot = (long)B * Nnew * cout;
  if (i >= tot) return;
  int c = (int)(i % cout);
  long t = i / cout;
  int n = (int)(t % Nnew);
  int b = (int)(t / Nnew);
  long ybase = (long)b * (7L * Nraw) * cout;
  float v;
  if (n < Nraw) {
    v = Y[ybase + (long)top[n] * cout + c];
  } else {
    int k = n - Nraw;
    v = 0.5f * (Y[ybase + (long)down[2 * k] * cout + c] +
                Y[ybase + (long)down[2 * k + 1] * cout + c]);
  }
  Xo[i] = (_Float16)v;
}

// ---------------------------------------------------------------------------
// Host orchestration
// ---------------------------------------------------------------------------
extern "C" void kernel_launch(void* const* d_in, const int* in_sizes, int n_in,
                              void* d_out, int out_size, void* d_ws, size_t ws_size,
                              hipStream_t stream) {
  (void)in_sizes; (void)n_in; (void)out_size; (void)ws_size;

  auto F = [&](int i) { return (const float*)d_in[i]; };
  auto I = [&](int i) { return (const int*)d_in[i]; };

  const int B = 64;
  const int NS[4]   = {162, 642, 2562, 10242};
  const int CIN[3]  = {512, 128, 64};
  const int COUT[3] = {128, 64, 32};
  const long OFF[4] = {0, 10368, 51456, 215424};  // map offsets within a hemisphere
  const long HEMI = 870912;                        // B * (162+642+2562+10242)
  const int S = 64;                                // stats slices

  // ---- workspace carve ----
  size_t off = 0;
  char* bp = (char*)d_ws;
  auto carve = [&](size_t bytes) -> void* {
    void* p = bp + off;
    off = (off + bytes + 255) & ~(size_t)255;
    return p;
  };
  _Float16* w1t = (_Float16*)carve(3584UL * 256 * 2);
  _Float16 *uwT[3], *cwT[3], *mw1T[3];
  for (int i = 0; i < 3; ++i) {
    uwT[i]  = (_Float16*)carve((size_t)CIN[i] * 7 * COUT[i] * 2);
    cwT[i]  = (_Float16*)carve((size_t)7 * COUT[i] * COUT[i] * 2);
    mw1T[i] = (_Float16*)carve((size_t)7 * COUT[i] * (COUT[i] / 2) * 2);
  }
  _Float16* x0h  = (_Float16*)carve(5308416UL * 2);   // input x as f16
  _Float16* xuph = (_Float16*)carve(20975616UL * 2);  // upconv-selected x, f16
  _Float16* xcvh = (_Float16*)carve(20975616UL * 2);  // post-conv x, f16
  float* hbuf = (float*)carve(10487808UL * 4);        // max B*N*ch (f32)
  float* vbuf = (float*)carve(655488UL * 4);          // max B*N
  float* ybuf = (float*)carve(36728832UL * 4);        // y buffer / raw conv out
  float* P    = (float*)carve(2UL * S * 512 * 4);
  float* ss   = (float*)carve(1024 * 4);

  // ---- launch helpers ----
  auto trans = [&](const float* W, _Float16* Wt, int K, int N) {
    long tot = (long)K * N;
    wtrans_kernel<<<dim3((unsigned)((tot + 255) / 256)), dim3(256), 0, stream>>>(W, Wt, K, N);
  };
  auto gemm = [&](const _Float16* X, const int* ng, const _Float16* Wt,
                  const float* bias, float* Out, long M, int N, int C, int Cout) {
    if (Cout % 64 == 0)
      wmma_gconv_kernel<4><<<dim3((unsigned)(M / 16), (unsigned)(Cout / 64)),
                             dim3(32), 0, stream>>>(X, ng, Wt, bias, Out, N, C, Cout);
    else if (Cout % 32 == 0)
      wmma_gconv_kernel<2><<<dim3((unsigned)(M / 16), (unsigned)(Cout / 32)),
                             dim3(32), 0, stream>>>(X, ng, Wt, bias, Out, N, C, Cout);
    else
      wmma_gconv_kernel<1><<<dim3((unsigned)(M / 16), (unsigned)(Cout / 16)),
                             dim3(32), 0, stream>>>(X, ng, Wt, bias, Out, N, C, Cout);
  };
  auto stats = [&](const float* X, long M, int C, const float* g, const float* be) {
    col_partial_kernel<<<dim3((unsigned)C, (unsigned)S), dim3(256), 0, stream>>>(X, M, C, P, S);
    col_finalize_kernel<<<dim3((unsigned)((C + 63) / 64)), dim3(64), 0, stream>>>(P, S, C, M, g, be, ss);
  };
  auto bnact = [&](const float* in, float* outf, _Float16* outh, long M, int C, int sigm) {
    long tot = M * C;
    bn_act_kernel<<<dim3((unsigned)((tot + 255) / 256)), dim3(256), 0, stream>>>(
        in, outf, outh, tot, C, ss, sigm);
  };
  auto dot1 = [&](const float* H, const int* ng, const float* w, const float* bptr,
                  float* out, long M, int N, int ch) {
    int lch = 31 - __builtin_clz((unsigned)ch);
    dot1_kernel<<<dim3((unsigned)(M / 8)), dim3(256), 0, stream>>>(H, ng, w, bptr, out, N, ch, lch);
  };

  // ---- input index map (recursive insertion-order flattening) ----
  // 0:x_l 1:x_r | 2..9 block_params: w1,b1,g1,be1,w2,b2,g2,be2
  // 10+14*i: uw,ub,cw,cb,cg,cbe, map{w1,b1,g1,be1,w2,b2,g2,be2}
  // 52..55: neigh_162/642/2562/10242 | 56..58: up_top | 59..61: up_down

  // ---- weight transposes (f32 -> f16, [K][N] -> [N][K]) ----
  trans(F(2), w1t, 3584, 256);
  for (int i = 0; i < 3; ++i) {
    int bi = 10 + 14 * i;
    trans(F(bi + 0), uwT[i],  CIN[i],       7 * COUT[i]);
    trans(F(bi + 2), cwT[i],  7 * COUT[i],  COUT[i]);
    trans(F(bi + 6), mw1T[i], 7 * COUT[i],  COUT[i] / 2);
  }

  // ---- per-hemisphere pipeline ----
  for (int hm = 0; hm < 2; ++hm) {
    const float* x0 = F(hm);
    float* outbase = (float*)d_out + (long)hm * HEMI;

    // input -> f16
    long nx0 = (long)B * NS[0] * 512;
    cvt16_kernel<<<dim3((unsigned)((nx0 + 255) / 256)), dim3(256), 0, stream>>>(x0, x0h, nx0);

    // level-0 attention block on raw input (C=512)
    long M0 = (long)B * NS[0];
    gemm(x0h, I(52), w1t, F(3), hbuf, M0, NS[0], 512, 256);
    stats(hbuf, M0, 256, F(4), F(5));
    bnact(hbuf, hbuf, nullptr, M0, 256, 0);
    dot1(hbuf, I(52), F(6), F(7), vbuf, M0, NS[0], 256);
    stats(vbuf, M0, 1, F(8), F(9));
    bnact(vbuf, outbase + OFF[0], nullptr, M0, 1, 1);

    const _Float16* xprevh = x0h;
    for (int i = 0; i < 3; ++i) {
      int bi = 10 + 14 * i;
      int cin = CIN[i], cout = COUT[i], ch = cout / 2;
      int Nraw = NS[i], Nnew = NS[i + 1];
      long Mr = (long)B * Nraw, Mn = (long)B * Nnew;
      const int* ng = I(53 + i);

      // upconv: y = x @ uw + ub   -> (B, Nraw*7, cout), f32
      gemm(xprevh, nullptr, uwT[i], F(bi + 1), ybuf, Mr, Nraw, cin, 7 * cout);
      // select top rows + mean of down pairs -> (B, Nnew, cout) f16
      long tot = (long)B * Nnew * cout;
      upsel_kernel<<<dim3((unsigned)((tot + 255) / 256)), dim3(256), 0, stream>>>(
          ybuf, I(56 + i), I(59 + i), xuph, B, Nraw, Nnew, cout);

      // gather-conv (raw f32 into ybuf, now free) + bn + lrelu -> new x (f16)
      gemm(xuph, ng, cwT[i], F(bi + 3), ybuf, Mn, Nnew, cout, cout);
      stats(ybuf, Mn, cout, F(bi + 4), F(bi + 5));
      bnact(ybuf, nullptr, xcvh, Mn, cout, 0);

      // attention: conv1 + bn + lrelu (f32, feeds scalar dot)
      gemm(xcvh, ng, mw1T[i], F(bi + 7), hbuf, Mn, Nnew, cout, ch);
      stats(hbuf, Mn, ch, F(bi + 8), F(bi + 9));
      bnact(hbuf, hbuf, nullptr, Mn, ch, 0);
      // attention: conv2 (Cout=1) + bn + lrelu + sigmoid -> map
      dot1(hbuf, ng, F(bi + 10), F(bi + 11), vbuf, Mn, Nnew, ch);
      stats(vbuf, Mn, 1, F(bi + 12), F(bi + 13));
      bnact(vbuf, outbase + OFF[i + 1], nullptr, Mn, 1, 1);

      xprevh = xcvh;
    }
  }
}
